// AngularSegmentationHeadv3_30013231464625
// MI455X (gfx1250) — matrix operand order repaired and checked by reading, not compiled, not run on hardware
//
#include <hip/hip_runtime.h>

// ---------------------------------------------------------------------------
// AngularSegmentationHead on MI455X (gfx1250).
//
// Roofline: mandatory HBM traffic = 512MiB (x, streamed) + 168MiB (result,
// streamed) ~= 714MB -> ~31us floor @ 23.3TB/s. GEMM work = 22.8 GFLOP; only
// the f16 WMMA pipe (v_wmma_f32_16x16x32_f16) keeps that far below the memory
// floor. All operands are swizzled in LDS so every WMMA fragment is fed by
// 2x ds_load_b128 (no scalar u16 gathers). Non-temporal global ops keep the
// 192MB L2 from being churned by the two streaming tensors.
// ---------------------------------------------------------------------------

#define HW_   (512 * 512)
#define NPIX  (8 * HW_)
#define PCLS  21
#define TILE  128               // pixels per block (8 waves x 16 rows)
#define XSTR  72                // sXh row stride in halves: 144B rows, 16B aligned
#define FSTR  68                // f32 buffer row stride in dwords (16B aligned rows)
#define CSTR  36                // cosine row stride in dwords (16B aligned rows)
#define SCALE_ 30.0f
#define COS_M_ 0.87758255958557128906f   // cos(0.5)
#define SIN_M_ 0.47942554950714111328f   // sin(0.5)

typedef __attribute__((ext_vector_type(16))) _Float16 v16h;
typedef __attribute__((ext_vector_type(8)))  _Float16 v8h;
typedef __attribute__((ext_vector_type(4)))  _Float16 v4h;
typedef __attribute__((ext_vector_type(8)))  float    v8f;
typedef __attribute__((ext_vector_type(4)))  float    v4f;

// Channel permutation: K-blocks of 8 stored in order 0,2,1,3,4,6,5,7 so the
// 16 halves each lane needs for an A fragment (K = {hv*8..+7} u {16+hv*8..+7}
// + 32ks, per ISA 7.12.2) are contiguous at half-offset 32*ks + 16*hv.
__device__ __forceinline__ int permc(int c) {
    int b  = c >> 3;
    int nb = (b & 4) | ((b & 1) << 1) | ((b >> 1) & 1);
    return nb * 8 + (c & 7);
}

// 32B contiguous, >=16B-aligned LDS fragment load -> 2x ds_load_b128.
__device__ __forceinline__ v16h load_frag(const _Float16* p) {
    union { v16h v; v8h h[2]; } u;
    u.h[0] = *(const v8h*)(p);
    u.h[1] = *(const v8h*)(p + 8);
    return u.v;
}

__global__ __launch_bounds__(256) void zero_loss_kernel(float* loss) { *loss = 0.0f; }

__global__ __launch_bounds__(256)
void angular_head_kernel(const float* __restrict__ x, const int* __restrict__ pred,
                         const float* __restrict__ conv_w, const float* __restrict__ conv_b,
                         const float* __restrict__ emb_w,
                         float* __restrict__ out, float* __restrict__ loss) {
    // B operands pre-swizzled to fragment order: [frag][lane][16 halves].
    __shared__ __align__(16) _Float16 sBconvF[8 * 512];   // conv (4 ntiles x 2 ksteps)
    __shared__ __align__(16) _Float16 sBenF[4 * 512];     // prototypes (2 ntiles x 2 ksteps)
    __shared__ __align__(16) float    sEn[PCLS * 64];     // normalized prototypes (commit)
    __shared__ __align__(16) float    sBias[64];
    __shared__ __align__(16) _Float16 sXh[TILE * XSTR];   // x tile f16 (reused for fx)
    __shared__ __align__(16) float    sF32[TILE * FSTR];  // xc f32; reused as cosine (CSTR)
    __shared__ float sLoss;

    const int tid  = threadIdx.x;
    const int lane = tid & 31;
    const int wave = tid >> 5;
    const long long n0 = (long long)blockIdx.x * TILE;
    const int b_  = (int)(n0 / HW_);     // 128 | HW -> tile never crosses images
    const int hw0 = (int)(n0 % HW_);

    if (tid == 0) sLoss = 0.0f;
    if (tid < 64) sBias[tid] = conv_b[tid];

    // conv_w (Cout,Cin) -> fragment-order f16: frag=nt*2+ks, per-lane contiguous.
    for (int i = tid; i < 8 * 512; i += 256) {
        int frag = i >> 9, ln = (i >> 4) & 31, e = i & 15;
        int nt = frag >> 1, ks = frag & 1;
        int K = ((ln >> 4) << 3) + (e & 7) + ((e >> 3) << 4) + 32 * ks;
        int N = nt * 16 + (ln & 15);
        sBconvF[i] = (_Float16)conv_w[N * 64 + K];
    }
    // normalized class prototypes (tiny: one thread per row)
    if (tid < PCLS) {
        float ss = 0.f;
#pragma unroll
        for (int c = 0; c < 64; ++c) { float v = emb_w[tid * 64 + c]; ss += v * v; }
        float inv = 1.0f / fmaxf(sqrtf(ss), 1e-12f);
#pragma unroll
        for (int c = 0; c < 64; ++c) sEn[tid * 64 + c] = emb_w[tid * 64 + c] * inv;
    }
    // x tile: streaming non-temporal b128 loads; channel-planar -> coalesced.
    for (int i4 = tid; i4 < TILE * 64 / 4; i4 += 256) {
        int p4 = (i4 & 31) * 4;
        int k  = i4 >> 5;
        v4f v = __builtin_nontemporal_load(
            (const v4f*)(x + (long long)b_ * 64 * HW_ + (long long)k * HW_ + hw0 + p4));
        int pos = permc(k);
        sXh[(p4 + 0) * XSTR + pos] = (_Float16)v[0];
        sXh[(p4 + 1) * XSTR + pos] = (_Float16)v[1];
        sXh[(p4 + 2) * XSTR + pos] = (_Float16)v[2];
        sXh[(p4 + 3) * XSTR + pos] = (_Float16)v[3];
    }
    __syncthreads();
    // prototypes -> fragment-order f16 B operand, N padded 21->32 with zeros
    for (int i = tid; i < 4 * 512; i += 256) {
        int frag = i >> 9, ln = (i >> 4) & 31, e = i & 15;
        int nt = frag >> 1, ks = frag & 1;
        int K = ((ln >> 4) << 3) + (e & 7) + ((e >> 3) << 4) + 32 * ks;
        int N = nt * 16 + (ln & 15);
        sBenF[i] = (N < PCLS) ? (_Float16)sEn[N * 64 + K] : (_Float16)0.0f;
    }
    __syncthreads();

    // ---- GEMM1: xc(128x64) = x(128x64) @ convW(64x64) ----
    {
        const int row0 = wave * 16;
        const int r    = row0 + (lane & 15);
        const int hv   = lane >> 4;
        v16h a0 = load_frag(&sXh[r * XSTR + 16 * hv]);        // ks=0
        v16h a1 = load_frag(&sXh[r * XSTR + 32 + 16 * hv]);   // ks=1
#pragma unroll
        for (int nt = 0; nt < 4; ++nt) {
            v8f acc = {};
            v16h b0 = load_frag(&sBconvF[(nt * 2 + 0) * 512 + lane * 16]);
            acc = __builtin_amdgcn_wmma_f32_16x16x32_f16(false, a0, false, b0,
                                                         (short)0, acc, false, false);
            v16h b1 = load_frag(&sBconvF[(nt * 2 + 1) * 512 + lane * 16]);
            acc = __builtin_amdgcn_wmma_f32_16x16x32_f16(false, a1, false, b1,
                                                         (short)0, acc, false, false);
            const int nc = nt * 16 + (lane & 15);
            const int mb = row0 + (hv << 3);                  // M = j + 8*(lane/16)
#pragma unroll
            for (int j = 0; j < 8; ++j) sF32[(mb + j) * FSTR + nc] = acc[j];
        }
    }
    __syncthreads();

    // ---- per-pixel: +bias, l2norm, fx -> f16 (reuse sXh), commitment loss ----
    int gt = 0;
    if (tid < TILE) {
        const int p = tid;
        gt = pred[n0 + p];
        v4f r[16];
        const v4f* xc4 = (const v4f*)(&sF32[p * FSTR]);
        const v4f* bb4 = (const v4f*)sBias;
        float ss = 0.f;
#pragma unroll
        for (int q = 0; q < 16; ++q) {
            v4f v = xc4[q] + bb4[q];
            r[q] = v;
            ss += v[0] * v[0] + v[1] * v[1] + v[2] * v[2] + v[3] * v[3];
        }
        float inv = 1.0f / fmaxf(sqrtf(ss), 1e-12f);
        const v4f* en4 = (const v4f*)(&sEn[gt * 64]);
        float commit = 0.f;
#pragma unroll
        for (int q = 0; q < 16; ++q) {
            v4f v = r[q];
            v4h h;
            h[0] = (_Float16)(v[0] * inv); h[1] = (_Float16)(v[1] * inv);
            h[2] = (_Float16)(v[2] * inv); h[3] = (_Float16)(v[3] * inv);
            int c0  = q * 4;
            int blk = c0 >> 3;
            int nb  = (blk & 4) | ((blk & 1) << 1) | ((blk >> 1) & 1);
            *(v4h*)(&sXh[p * XSTR + nb * 8 + (c0 & 7)]) = h;   // ds_store_b64
            v4f e = en4[q];
            v4f d = e - v;
            commit += d[0] * d[0] + d[1] * d[1] + d[2] * d[2] + d[3] * d[3];
        }
        atomicAdd(&sLoss, commit * (1.0f / ((float)NPIX * 64.0f)));
    }
    __syncthreads();

    // ---- GEMM2: cosine(128x32pad) = fx(128x64) @ enT(64x32) ----
    {
        const int row0 = wave * 16;
        const int r    = row0 + (lane & 15);
        const int hv   = lane >> 4;
        v16h a0 = load_frag(&sXh[r * XSTR + 16 * hv]);
        v16h a1 = load_frag(&sXh[r * XSTR + 32 + 16 * hv]);
#pragma unroll
        for (int nt = 0; nt < 2; ++nt) {
            v8f acc = {};
            v16h b0 = load_frag(&sBenF[(nt * 2 + 0) * 512 + lane * 16]);
            acc = __builtin_amdgcn_wmma_f32_16x16x32_f16(false, a0, false, b0,
                                                         (short)0, acc, false, false);
            v16h b1 = load_frag(&sBenF[(nt * 2 + 1) * 512 + lane * 16]);
            acc = __builtin_amdgcn_wmma_f32_16x16x32_f16(false, a1, false, b1,
                                                         (short)0, acc, false, false);
            const int nc = nt * 16 + (lane & 15);
            const int mb = row0 + (hv << 3);
#pragma unroll
            for (int j = 0; j < 8; ++j) sF32[(mb + j) * CSTR + nc] = acc[j];
        }
    }
    __syncthreads();

    // ---- per-pixel: softmax over 21 real classes + ArcFace CE ----
    if (tid < TILE) {
        const int p = tid;
        float cs[PCLS], pe[PCLS];
        float m = -1e30f;
#pragma unroll
        for (int c = 0; c < PCLS; ++c) { cs[c] = sF32[p * CSTR + c]; m = fmaxf(m, cs[c]); }
        float s = 0.f;
#pragma unroll
        for (int c = 0; c < PCLS; ++c) { pe[c] = __expf(cs[c] - m); s += pe[c]; }
        float invs = 1.0f / s;
#pragma unroll
        for (int c = 0; c < PCLS; ++c) sF32[p * CSTR + c] = pe[c] * invs;

        float cg  = cs[gt];
        float sg  = sqrtf(fminf(fmaxf(1.0f - cg * cg, 0.0f), 1.0f));
        float phi = cg * COS_M_ - sg * SIN_M_;
        phi = (cg > 0.0f) ? phi : cg;                 // easy_margin
        float ph = (float)(_Float16)phi;              // fp16 round-trip per reference
        float ng = cg * ph;
        float pos = __expf(SCALE_ * ng);
        float sa  = 0.f;
#pragma unroll
        for (int c = 0; c < PCLS; ++c) {
            float l = (c == gt) ? SCALE_ * ng : SCALE_ * cs[c];
            sa += __expf(l);
        }
        float ce = -__logf(pos / (sa + 1e-7f) + 1e-7f);
        atomicAdd(&sLoss, ce * (1.0f / (float)NPIX));
    }
    __syncthreads();

    // ---- class-planar probability writes: non-temporal b128, coalesced ----
    for (int i = tid; i < PCLS * 32; i += 256) {
        int cls = i >> 5;
        int p4  = (i & 31) * 4;
        v4f v;
        v[0] = sF32[(p4 + 0) * CSTR + cls];
        v[1] = sF32[(p4 + 1) * CSTR + cls];
        v[2] = sF32[(p4 + 2) * CSTR + cls];
        v[3] = sF32[(p4 + 3) * CSTR + cls];
        __builtin_nontemporal_store(
            v, (v4f*)(out + (long long)b_ * PCLS * HW_ + (long long)cls * HW_ + hw0 + p4));
    }
    if (tid == 0) atomicAdd(loss, sLoss);
}

extern "C" void kernel_launch(void* const* d_in, const int* in_sizes, int n_in,
                              void* d_out, int out_size, void* d_ws, size_t ws_size,
                              hipStream_t stream) {
    (void)in_sizes; (void)n_in; (void)d_ws; (void)ws_size;
    const float* x      = (const float*)d_in[0];
    const int*   pred   = (const int*)d_in[1];   // integer input -> const int* per harness ABI
    const float* conv_w = (const float*)d_in[2];
    const float* conv_b = (const float*)d_in[3];
    const float* emb_w  = (const float*)d_in[4];
    float* out  = (float*)d_out;
    float* loss = out + (out_size - 1);          // scalar loss appended after result tensor

    hipLaunchKernelGGL(zero_loss_kernel, dim3(1), dim3(1), 0, stream, loss);
    hipLaunchKernelGGL(angular_head_kernel, dim3(NPIX / TILE), dim3(256), 0, stream,
                       x, pred, conv_w, conv_b, emb_w, out, loss);
}
